// LinearMixtureOfMixers_75007308857796
// MI455X (gfx1250) — matrix operand order repaired and checked by reading, not compiled, not run on hardware
//
#include <hip/hip_runtime.h>
#include <hip/hip_bf16.h>

// ---------------------------------------------------------------------------
// LinearMixtureOfMixers on MI455X (gfx1250, wave32, WMMA)
// B=8, N=512, D=1024, E=8, H=8, TOP_K=2, HD=128
//
// v3: ping-pong double buffering (2x unrolled K-loop) removes the v_mov
// hand-off block and its hazard NOPs; inner loop is loads + v_wmma only.
// ---------------------------------------------------------------------------

#define B_  8
#define N_  512
#define D_  1024
#define E_  8
#define H_  8
#define HD_ 128

typedef __bf16 bf16_t;
typedef __attribute__((ext_vector_type(16))) __bf16 bf16x16;
typedef __attribute__((ext_vector_type(8)))  __bf16 bf16x8;
typedef __attribute__((ext_vector_type(8)))  float  v8f;

// ---- WMMA fragment helpers -------------------------------------------------
// A-matrix 16x32 bf16 layout (ISA 7.12.2): lane<16 -> M=lane, K in {0..7}u{16..23};
// lane>=16 -> same M, K in {8..15}u{24..31}.
__device__ __forceinline__ bf16x16 frag_a_bf16(const bf16_t* __restrict__ row,
                                               int kb, int lane) {
  const int half = (lane >> 4) << 3;
  bf16x8 lo = *reinterpret_cast<const bf16x8*>(row + kb + half);
  bf16x8 hi = *reinterpret_cast<const bf16x8*>(row + kb + 16 + half);
  bf16x16 f;
#pragma unroll
  for (int i = 0; i < 8; ++i) { f[i] = lo[i]; f[8 + i] = hi[i]; }
  return f;
}

// B-matrix 32x16 bf16: lanes 0-15 carry K={0..7}u{16..23} of column (lane&15),
// lanes 16-31 carry K={8..15}u{24..31}.
__device__ __forceinline__ bf16x16 frag_b_bf16(const bf16_t* __restrict__ row,
                                               int kb, int lane) {
  const int half = (lane >> 4) << 3;
  bf16x8 lo = *reinterpret_cast<const bf16x8*>(row + kb + half);
  bf16x8 hi = *reinterpret_cast<const bf16x8*>(row + kb + 16 + half);
  bf16x16 f;
#pragma unroll
  for (int i = 0; i < 8; ++i) { f[i] = lo[i]; f[8 + i] = hi[i]; }
  return f;
}

#define WMMA_BF16(a, b, c)                                                   \
  __builtin_amdgcn_wmma_f32_16x16x32_bf16(false, (a), false, (b), (short)0,  \
                                          (c), false, false)

// ---------------------------------------------------------------------------
// Elementwise fp32 -> bf16 (8 elems/thread, vectorized)
// ---------------------------------------------------------------------------
__global__ __launch_bounds__(256) void f32_to_bf16_kernel(
    const float* __restrict__ s, bf16_t* __restrict__ d, int n8) {
  const int i = blockIdx.x * 256 + threadIdx.x;
  if (i >= n8) return;
  const float4* p = reinterpret_cast<const float4*>(s) + (size_t)i * 2;
  float4 a = p[0], b = p[1];
  bf16x8 o;
  o[0] = (bf16_t)a.x; o[1] = (bf16_t)a.y; o[2] = (bf16_t)a.z; o[3] = (bf16_t)a.w;
  o[4] = (bf16_t)b.x; o[5] = (bf16_t)b.y; o[6] = (bf16_t)b.z; o[7] = (bf16_t)b.w;
  reinterpret_cast<bf16x8*>(d)[i] = o;
}

// ---------------------------------------------------------------------------
// GEMM: C[M,Nc] = A[M,K] @ W[Nc,K]^T + bias   (bf16 operands, fp32 out)
// One wave owns a 32x64 strip: 2 A-frags x 4 B-frags -> 8 WMMAs per K-step.
// Ping-pong double buffering, K % 64 == 0 assumed.
// grid = (M/32, Nc/(64*8)); block = 256 (8 waves).
// ---------------------------------------------------------------------------
struct FragsG { bf16x16 a0, a1, b0, b1, b2, b3; };

__device__ __forceinline__ FragsG load_frags_g(
    const bf16_t* ar0, const bf16_t* ar1, const bf16_t* w0, const bf16_t* w1,
    const bf16_t* w2, const bf16_t* w3, int kb, int lane) {
  FragsG f;
  f.a0 = frag_a_bf16(ar0, kb, lane); f.a1 = frag_a_bf16(ar1, kb, lane);
  f.b0 = frag_b_bf16(w0, kb, lane);  f.b1 = frag_b_bf16(w1, kb, lane);
  f.b2 = frag_b_bf16(w2, kb, lane);  f.b3 = frag_b_bf16(w3, kb, lane);
  return f;
}

#define GEMM_WMMA8(F)                                   \
  do {                                                  \
    acc[0][0] = WMMA_BF16((F).a0, (F).b0, acc[0][0]);   \
    acc[0][1] = WMMA_BF16((F).a0, (F).b1, acc[0][1]);   \
    acc[0][2] = WMMA_BF16((F).a0, (F).b2, acc[0][2]);   \
    acc[0][3] = WMMA_BF16((F).a0, (F).b3, acc[0][3]);   \
    acc[1][0] = WMMA_BF16((F).a1, (F).b0, acc[1][0]);   \
    acc[1][1] = WMMA_BF16((F).a1, (F).b1, acc[1][1]);   \
    acc[1][2] = WMMA_BF16((F).a1, (F).b2, acc[1][2]);   \
    acc[1][3] = WMMA_BF16((F).a1, (F).b3, acc[1][3]);   \
  } while (0)

__global__ __launch_bounds__(256) void wmma_gemm_nt(
    const bf16_t* __restrict__ A, const bf16_t* __restrict__ W,
    const float* __restrict__ bias, float* __restrict__ C,
    int M, int Nc, int K) {
  const int lane  = threadIdx.x & 31;
  const int wave  = threadIdx.x >> 5;
  const int mtile = blockIdx.x;                 // M/32
  const int ntile = blockIdx.y * 8 + wave;      // Nc/64
  const int msub  = lane & 15;
  const int col0  = ntile * 64;

  const bf16_t* ar0 = A + (size_t)(mtile * 32 + msub) * K;
  const bf16_t* ar1 = ar0 + (size_t)16 * K;
  const bf16_t* w0 = W + (size_t)(col0 +  0 + msub) * K;
  const bf16_t* w1 = W + (size_t)(col0 + 16 + msub) * K;
  const bf16_t* w2 = W + (size_t)(col0 + 32 + msub) * K;
  const bf16_t* w3 = W + (size_t)(col0 + 48 + msub) * K;

  v8f acc[2][4] = {};

  FragsG f0 = load_frags_g(ar0, ar1, w0, w1, w2, w3, 0, lane);
  for (int kb = 0; kb < K; kb += 64) {
    FragsG f1 = load_frags_g(ar0, ar1, w0, w1, w2, w3, kb + 32, lane);
    __builtin_prefetch(ar0 + kb + 64, 0, 1);    // global_prefetch next A tile
    GEMM_WMMA8(f0);
    if (kb + 64 < K)                            // uniform branch
      f0 = load_frags_g(ar0, ar1, w0, w1, w2, w3, kb + 64, lane);
    GEMM_WMMA8(f1);
  }

  // C/D 16x16 layout: lane<16 -> M=r, lane>=16 -> M=r+8; N = lane&15.
  const int nhalf = lane >> 4;
#pragma unroll
  for (int j = 0; j < 4; ++j) {
    const int col = col0 + j * 16 + msub;
    const float bv = bias[col];
#pragma unroll
    for (int i = 0; i < 2; ++i) {
#pragma unroll
      for (int r = 0; r < 8; ++r) {
        const int m = mtile * 32 + i * 16 + r + 8 * nhalf;
        C[(size_t)m * Nc + col] = acc[i][j][r] + bv;
      }
    }
  }
}

// ---------------------------------------------------------------------------
// Column mean over n:  xmean[b,d] = mean_n x[b,n,d]
// ---------------------------------------------------------------------------
__global__ __launch_bounds__(256) void colmean_kernel(const float* __restrict__ x,
                                                      float* __restrict__ xmean) {
  const int b = blockIdx.x;
  for (int d = threadIdx.x; d < D_; d += blockDim.x) {
    const float* p = x + (size_t)b * N_ * D_ + d;
    float s = 0.f;
    for (int n = 0; n < N_; ++n) s += p[(size_t)n * D_];
    xmean[b * D_ + d] = s * (1.0f / N_);
  }
}

// ---------------------------------------------------------------------------
// Router: logits, softmax, top-2 (normalized), aux loss. One block, 64 thr.
// ---------------------------------------------------------------------------
__global__ __launch_bounds__(64) void router_kernel(
    const float* __restrict__ xmean, const float* __restrict__ rw,
    int* __restrict__ topi, float* __restrict__ topw, float* __restrict__ aux) {
  __shared__ float logits[B_][E_];
  __shared__ float probs[B_][E_];
  __shared__ int top1[B_];
  const int t = threadIdx.x;
  {
    const int b = t >> 3, e = t & 7;
    float s = 0.f;
    for (int d = 0; d < D_; ++d) s += xmean[b * D_ + d] * rw[e * D_ + d];
    logits[b][e] = s;
  }
  __syncthreads();
  if (t < B_) {
    const int b = t;
    float mx = logits[b][0];
    for (int e = 1; e < E_; ++e) mx = fmaxf(mx, logits[b][e]);
    float p[E_], s = 0.f;
    for (int e = 0; e < E_; ++e) { p[e] = __expf(logits[b][e] - mx); s += p[e]; }
    const float inv = 1.f / s;
    for (int e = 0; e < E_; ++e) { p[e] *= inv; probs[b][e] = p[e]; }
    int e0 = 0;
    for (int e = 1; e < E_; ++e) if (p[e] > p[e0]) e0 = e;
    int e1 = (e0 == 0) ? 1 : 0;
    for (int e = 0; e < E_; ++e) if (e != e0 && p[e] > p[e1]) e1 = e;
    const float wn = 1.f / (p[e0] + p[e1]);
    topi[b * 2] = e0; topi[b * 2 + 1] = e1;
    topw[b * 2] = p[e0] * wn; topw[b * 2 + 1] = p[e1] * wn;
    top1[b] = e0;
  }
  __syncthreads();
  if (t == 0) {
    float a = 0.f;
    for (int e = 0; e < E_; ++e) {
      float mp = 0.f, mm = 0.f;
      for (int b = 0; b < B_; ++b) {
        mp += probs[b][e];
        mm += (top1[b] == e) ? 1.f : 0.f;
      }
      a += (mp * (1.f / B_)) * (mm * (1.f / B_));
    }
    aux[0] = (float)E_ * a;
  }
}

// ---------------------------------------------------------------------------
// Row softmax of mixing weights -> bf16.  rows = E*H*N, width N=512.
// One wave per row; grid = rows/8, block = 256.
// ---------------------------------------------------------------------------
__global__ __launch_bounds__(256) void softmax_rows_kernel(
    const float* __restrict__ w, bf16_t* __restrict__ o) {
  const int row  = blockIdx.x * 8 + (threadIdx.x >> 5);
  const int lane = threadIdx.x & 31;
  const float* src = w + (size_t)row * N_;
  float v[16];
  float mx = -3.4e38f;
#pragma unroll
  for (int i = 0; i < 16; ++i) { v[i] = src[lane + 32 * i]; mx = fmaxf(mx, v[i]); }
#pragma unroll
  for (int off = 16; off > 0; off >>= 1) mx = fmaxf(mx, __shfl_xor(mx, off, 32));
  float s = 0.f;
#pragma unroll
  for (int i = 0; i < 16; ++i) { v[i] = __expf(v[i] - mx); s += v[i]; }
#pragma unroll
  for (int off = 16; off > 0; off >>= 1) s += __shfl_xor(s, off, 32);
  const float inv = 1.f / s;
  bf16_t* dst = o + (size_t)row * N_;
#pragma unroll
  for (int i = 0; i < 16; ++i) dst[lane + 32 * i] = (bf16_t)(v[i] * inv);
}

// ---------------------------------------------------------------------------
// LayerNorm across n of xp viewed as [b,h,d,n]; emit bf16 rows (n-contiguous).
// grid = B*H blocks, 128 threads (thread = d; coalesced across d).
// ---------------------------------------------------------------------------
__global__ __launch_bounds__(128) void layernorm_kernel(
    const float* __restrict__ xp, bf16_t* __restrict__ xn) {
  const int bh = blockIdx.x;
  const int b = bh >> 3, h = bh & 7;
  const int d = threadIdx.x;
  const float* col = xp + (size_t)b * N_ * D_ + h * HD_ + d;
  float s = 0.f, s2 = 0.f;
  for (int n = 0; n < N_; ++n) {
    const float v = col[(size_t)n * D_];
    s += v; s2 += v * v;
  }
  const float mu = s * (1.f / N_);
  const float var = s2 * (1.f / N_) - mu * mu;
  const float rs = rsqrtf(var + 1e-5f);
  bf16_t* dst = xn + ((size_t)bh * HD_ + d) * N_;
  for (int n = 0; n < N_; ++n)
    dst[n] = (bf16_t)((col[(size_t)n * D_] - mu) * rs);
}

// ---------------------------------------------------------------------------
// Top-2 mixing einsum, bf16 output for the final GEMM.
//   mixedb[b, m, h*HD+d] = sum_k topw[b,k]*( xn[b,h] @ wsm[e_k,h]^T )[d,m]
//                        + sum_k topw[b,k]*bias[e_k,h,m]
// One wave owns 32(d) x 32(m): 2 A-frags shared across both experts,
// 4 B-frags -> 8 WMMAs per K-step, ping-pong double buffered.
// grid = (N/64, B*H); block = 256 (8 waves: w&3 -> d32, w>>2 -> m32 pair).
// ---------------------------------------------------------------------------
struct FragsM { bf16x16 a0, a1, p0, p1, q0, q1; };

__device__ __forceinline__ FragsM load_frags_m(
    const bf16_t* ar0, const bf16_t* ar1, const bf16_t* p0r, const bf16_t* p1r,
    const bf16_t* q0r, const bf16_t* q1r, int kb, int lane) {
  FragsM f;
  f.a0 = frag_a_bf16(ar0, kb, lane); f.a1 = frag_a_bf16(ar1, kb, lane);
  f.p0 = frag_b_bf16(p0r, kb, lane); f.p1 = frag_b_bf16(p1r, kb, lane);
  f.q0 = frag_b_bf16(q0r, kb, lane); f.q1 = frag_b_bf16(q1r, kb, lane);
  return f;
}

#define MIX_WMMA8(F)                                          \
  do {                                                        \
    acc[0][0][0] = WMMA_BF16((F).a0, (F).p0, acc[0][0][0]);   \
    acc[0][1][0] = WMMA_BF16((F).a0, (F).p1, acc[0][1][0]);   \
    acc[0][0][1] = WMMA_BF16((F).a0, (F).q0, acc[0][0][1]);   \
    acc[0][1][1] = WMMA_BF16((F).a0, (F).q1, acc[0][1][1]);   \
    acc[1][0][0] = WMMA_BF16((F).a1, (F).p0, acc[1][0][0]);   \
    acc[1][1][0] = WMMA_BF16((F).a1, (F).p1, acc[1][1][0]);   \
    acc[1][0][1] = WMMA_BF16((F).a1, (F).q0, acc[1][0][1]);   \
    acc[1][1][1] = WMMA_BF16((F).a1, (F).q1, acc[1][1][1]);   \
  } while (0)

__global__ __launch_bounds__(256) void wmma_mix_kernel(
    const bf16_t* __restrict__ xn,     // [B,H,HD,N]
    const bf16_t* __restrict__ wsm,    // [E,H,N,N] (softmaxed, bf16)
    const float*  __restrict__ bias,   // [E,H,N]
    const int*    __restrict__ topi,   // [B,2]
    const float*  __restrict__ topw,   // [B,2]
    bf16_t*       __restrict__ mixedb) // [B,N,D] bf16
{
  const int lane = threadIdx.x & 31;
  const int wave = threadIdx.x >> 5;
  const int d32  = wave & 3;                       // HD/32 = 4
  const int m32  = blockIdx.x * 2 + (wave >> 2);   // N/32 = 16
  const int bh   = blockIdx.y;                     // B*H
  const int b = bh >> 3, h = bh & 7;
  const int msub = lane & 15;

  const int e0 = topi[b * 2], e1 = topi[b * 2 + 1];
  const float t0 = topw[b * 2], t1 = topw[b * 2 + 1];

  const bf16_t* ar0 = xn + (size_t)(bh * HD_ + d32 * 32 + msub) * N_;
  const bf16_t* ar1 = ar0 + (size_t)16 * N_;
  const bf16_t* p0r = wsm + (size_t)((e0 * H_ + h) * N_ + m32 * 32 + msub) * N_;
  const bf16_t* p1r = p0r + (size_t)16 * N_;
  const bf16_t* q0r = wsm + (size_t)((e1 * H_ + h) * N_ + m32 * 32 + msub) * N_;
  const bf16_t* q1r = q0r + (size_t)16 * N_;

  v8f acc[2][2][2] = {};   // [dpair][mpair][expert]

  FragsM f0 = load_frags_m(ar0, ar1, p0r, p1r, q0r, q1r, 0, lane);
  for (int kb = 0; kb < N_; kb += 64) {
    FragsM f1 = load_frags_m(ar0, ar1, p0r, p1r, q0r, q1r, kb + 32, lane);
    MIX_WMMA8(f0);
    if (kb + 64 < N_)                            // uniform branch
      f0 = load_frags_m(ar0, ar1, p0r, p1r, q0r, q1r, kb + 64, lane);
    MIX_WMMA8(f1);
  }

  const int nhalf = lane >> 4;
#pragma unroll
  for (int j = 0; j < 2; ++j) {
    const int m = m32 * 32 + j * 16 + msub;        // token index (C column)
    const float bmix = t0 * bias[(e0 * H_ + h) * N_ + m]
                     + t1 * bias[(e1 * H_ + h) * N_ + m];
#pragma unroll
    for (int i = 0; i < 2; ++i) {
      bf16x8 o;
#pragma unroll
      for (int r = 0; r < 8; ++r)
        o[r] = (bf16_t)(t0 * acc[i][j][0][r] + t1 * acc[i][j][1][r] + bmix);
      // rows d contiguous in mixedb[b,m,h*HD+d]
      const int dbase = d32 * 32 + i * 16 + 8 * nhalf;
      bf16_t* dst = mixedb + ((size_t)b * N_ + m) * D_ + h * HD_ + dbase;
      *reinterpret_cast<bf16x8*>(dst) = o;
    }
  }
}

// ---------------------------------------------------------------------------
// Host launcher
// ---------------------------------------------------------------------------
extern "C" void kernel_launch(void* const* d_in, const int* in_sizes, int n_in,
                              void* d_out, int out_size, void* d_ws, size_t ws_size,
                              hipStream_t stream) {
  const float* x        = (const float*)d_in[0];  // [B,N,D]
  const float* weight   = (const float*)d_in[1];  // [E,H,N,N]
  const float* bias     = (const float*)d_in[2];  // [E,H,N]
  const float* router_w = (const float*)d_in[3];  // [E,D]
  const float* in_w     = (const float*)d_in[4];  // [D,D]
  const float* in_b     = (const float*)d_in[5];  // [D]
  const float* out_w    = (const float*)d_in[6];  // [D,D]
  const float* out_b    = (const float*)d_in[7];  // [D]
  float* out = (float*)d_out;                     // [B*N*D] ++ [aux]

  // Workspace layout (256B-aligned slabs), ~77 MB total.
  char* ws = (char*)d_ws;
  size_t off = 0;
  auto alloc = [&](size_t bytes) {
    void* p = ws + off;
    off = (off + bytes + 255) & ~(size_t)255;
    return p;
  };
  float*  xp     = (float*) alloc((size_t)B_ * N_ * D_ * 4);       // 16 MB
  bf16_t* xb     = (bf16_t*)alloc((size_t)B_ * N_ * D_ * 2);       //  8 MB
  bf16_t* mixedb = (bf16_t*)alloc((size_t)B_ * N_ * D_ * 2);       //  8 MB
  bf16_t* xn     = (bf16_t*)alloc((size_t)B_ * H_ * HD_ * N_ * 2); //  8 MB
  bf16_t* wsm    = (bf16_t*)alloc((size_t)E_ * H_ * N_ * N_ * 2);  // 32 MB
  bf16_t* inwb   = (bf16_t*)alloc((size_t)D_ * D_ * 2);            //  2 MB
  bf16_t* outwb  = (bf16_t*)alloc((size_t)D_ * D_ * 2);            //  2 MB
  float*  xmean  = (float*) alloc((size_t)B_ * D_ * 4);
  int*    topi   = (int*)   alloc((size_t)B_ * 2 * 4);
  float*  topw   = (float*) alloc((size_t)B_ * 2 * 4);

  // Router path (tiny).
  colmean_kernel<<<B_, 256, 0, stream>>>(x, xmean);
  router_kernel<<<1, 64, 0, stream>>>(xmean, router_w, topi, topw,
                                      out + (size_t)B_ * N_ * D_);

  // One-time operand conversions to bf16.
  {
    const int nx8 = (B_ * N_ * D_) / 8;   // 524288
    const int nw8 = (D_ * D_) / 8;        // 131072
    f32_to_bf16_kernel<<<(nx8 + 255) / 256, 256, 0, stream>>>(x, xb, nx8);
    f32_to_bf16_kernel<<<(nw8 + 255) / 256, 256, 0, stream>>>(in_w, inwb, nw8);
    f32_to_bf16_kernel<<<(nw8 + 255) / 256, 256, 0, stream>>>(out_w, outwb, nw8);
  }

  // xp = x @ in_w^T + in_b   (M=4096, Nc=1024, K=1024)
  wmma_gemm_nt<<<dim3((B_ * N_) / 32, 2), 256, 0, stream>>>(
      xb, inwb, in_b, xp, B_ * N_, D_, D_);

  // Softmax all expert mixing rows -> bf16.
  softmax_rows_kernel<<<(E_ * H_ * N_) / 8, 256, 0, stream>>>(weight, wsm);

  // LayerNorm over n -> bf16 xn rows.
  layernorm_kernel<<<B_ * H_, 128, 0, stream>>>(xp, xn);

  // Top-2 mixing einsum (+bias, +combine) -> mixedb[b,n,D] (bf16).
  wmma_mix_kernel<<<dim3(N_ / 64, B_ * H_), 256, 0, stream>>>(
      xn, wsm, bias, topi, topw, mixedb);

  // out = mixed @ out_w^T + out_b
  wmma_gemm_nt<<<dim3((B_ * N_) / 32, 2), 256, 0, stream>>>(
      mixedb, outwb, out_b, out, B_ * N_, D_, D_);
}
